// RNNModel_67379446940230
// MI455X (gfx1250) — compile-verified, matrix operand order checked by reading, not run on hardware
//
#include <hip/hip_runtime.h>
#include <hip/hip_bf16.h>
#include <stdint.h>

// ---------------------------------------------------------------------------
// CDNA5 (gfx1250) RNN: WMMA f32_16x16x32_f16.
//  - xp stored in packed WMMA C-fragment order -> vector ld/st of accumulators
//  - recurrence: 1 WGP, h double-buffered in LDS, 1 barrier/step
//  - W_hh pre-packed into per-lane B-fragments (coalesced 32B/lane)
// ---------------------------------------------------------------------------

typedef __attribute__((ext_vector_type(16))) _Float16 v16h;
typedef __attribute__((ext_vector_type(8)))  _Float16 v8h;
typedef __attribute__((ext_vector_type(8)))  float    v8f;

union FragU { v16h v; v8h h[2]; };

// Load a 16x32 (f16) A/B WMMA fragment from an LDS row (row-major, K-contig).
// Lane layout per ISA 7.12.2: lanes 0-15 hold K chunks {0..7,16..23},
// lanes 16-31 hold K chunks {8..15,24..31}; halves e0..7 = chunk0, e8..15 = chunk1.
__device__ __forceinline__ v16h load_frag_lds(const _Float16* rowBase, int laneHi) {
  FragU u;
  u.h[0] = *(const v8h*)(rowBase + laneHi * 8);
  u.h[1] = *(const v8h*)(rowBase + laneHi * 8 + 16);
  return u.v;
}

// Packed xp layout: [t][ntile(32)][mt(4)][laneHi(2)][lane15(16)][v(8)]
// One timestep slice = 32768 floats (128KB, contiguous).
__device__ __forceinline__ size_t xp_base(int t, int ntile, int mt,
                                          int laneHi, int lane15) {
  return (size_t)t * 32768 + ntile * 1024 + mt * 256 + laneHi * 128 + lane15 * 8;
}

// ---------------------------------------------------------------------------
// Projection GEMM: Cpk = A'[M][K] * W[N][K]^T + (b0 + b1), f32 accumulate.
// M-dimension is t-major (m = t*64 + b). If PERM_A, A source is b-major
// [B][T][K] and row m reads source row (m&63)*1024 + (m>>6).
// Output written in packed fragment order (see xp_base).
// Block = 256 threads (8 waves), block tile 128(M) x 64(N), k-slice = 32.
// ---------------------------------------------------------------------------
template <typename AT, bool PERM_A>
__global__ __launch_bounds__(256) void proj_gemm(
    const AT* __restrict__ A, const float* __restrict__ W,
    const float* __restrict__ b0, const float* __restrict__ b1,
    float* __restrict__ Cpk, int M, int N, int K)
{
  __shared__ _Float16 As[128 * 40];  // stride 40 halves (80B, 16B aligned)
  __shared__ _Float16 Bs[64 * 40];

  const int tid    = threadIdx.x;
  const int lane   = tid & 31;
  const int wave   = tid >> 5;      // 0..7
  const int laneHi = lane >> 4;
  const int lane15 = lane & 15;
  const int n0 = blockIdx.x * 64;
  const int m0 = blockIdx.y * 128;
  const int wm = wave >> 1;         // 0..3  -> 32-row wave tile
  const int wn = wave & 1;          // 0..1  -> 32-col wave tile

  v8f acc[2][2];
#pragma unroll
  for (int tm = 0; tm < 2; ++tm)
#pragma unroll
    for (int tn = 0; tn < 2; ++tn) {
      const int n = n0 + wn * 32 + tn * 16 + lane15;
      const float bv = b0[n] + b1[n];
#pragma unroll
      for (int v = 0; v < 8; ++v) acc[tm][tn][v] = bv;
    }

  for (int k0 = 0; k0 < K; k0 += 32) {
    // Stage A slab (128x32) with convert-to-f16, coalesced along K.
#pragma unroll
    for (int i = 0; i < 16; ++i) {
      const int idx = i * 256 + tid;
      const int r = idx >> 5, c = idx & 31;
      const int m = m0 + r;
      const size_t srcRow = PERM_A ? ((size_t)(m & 63) * 1024 + (m >> 6))
                                   : (size_t)m;
      As[r * 40 + c] = (_Float16)(float)A[srcRow * K + k0 + c];
    }
    // Stage W slab (64x32).
#pragma unroll
    for (int i = 0; i < 8; ++i) {
      const int idx = i * 256 + tid;
      const int r = idx >> 5, c = idx & 31;
      Bs[r * 40 + c] = (_Float16)W[(size_t)(n0 + r) * K + k0 + c];
    }
    if (k0 + 32 < K) {  // hint next W slab into cache: global_prefetch_b8
      __builtin_prefetch(&W[(size_t)(n0 + (tid >> 2)) * K + k0 + 32], 0, 1);
    }
    __syncthreads();

    v16h bfrag[2];
#pragma unroll
    for (int tn = 0; tn < 2; ++tn)
      bfrag[tn] = load_frag_lds(&Bs[(wn * 32 + tn * 16 + lane15) * 40], laneHi);
#pragma unroll
    for (int tm = 0; tm < 2; ++tm) {
      v16h afrag = load_frag_lds(&As[(wm * 32 + tm * 16 + lane15) * 40], laneHi);
#pragma unroll
      for (int tn = 0; tn < 2; ++tn)
        acc[tm][tn] = __builtin_amdgcn_wmma_f32_16x16x32_f16(
            false, afrag, false, bfrag[tn], (short)0, acc[tm][tn], false, false);
    }
    __syncthreads();
  }

  // Epilogue: each 16x16 tile sits inside one timestep's 64-row block, so the
  // whole v8f accumulator goes out as one 32B vector store in packed order.
#pragma unroll
  for (int tm = 0; tm < 2; ++tm) {
    const int tileM = m0 + wm * 32 + tm * 16;
    const int t  = tileM >> 6;
    const int mt = (tileM & 63) >> 4;
#pragma unroll
    for (int tn = 0; tn < 2; ++tn) {
      const int ntile = (n0 >> 4) + wn * 2 + tn;
      *(v8f*)(Cpk + xp_base(t, ntile, mt, laneHi, lane15)) = acc[tm][tn];
    }
  }
}

// ---------------------------------------------------------------------------
// Pre-pack W_hh (f32 [512][512]) into per-lane WMMA B-fragments (f16) so the
// recurrence kernel reads perfectly coalesced 32B/lane fragments.
// Fragment id = ntile*16 + ks; buffer = [frag][lane][16 halves].
// ---------------------------------------------------------------------------
__global__ __launch_bounds__(256) void prepack_whh(
    const float* __restrict__ W, _Float16* __restrict__ Wpk)
{
  const int gid    = blockIdx.x * 256 + threadIdx.x;  // 0..16383
  const int lane   = gid & 31;
  const int ks     = (gid >> 5) & 15;
  const int ntile  = gid >> 9;                        // 0..31
  const int laneHi = lane >> 4;
  const int n      = ntile * 16 + (lane & 15);
  _Float16* dst = Wpk + (size_t)gid * 16;
#pragma unroll
  for (int e = 0; e < 16; ++e) {
    int kloc;
    if (!laneHi) kloc = (e < 8) ? e        : (e + 8);   // {0..7,16..23}
    else         kloc = (e < 8) ? (e + 8)  : (e + 16);  // {8..15,24..31}
    dst[e] = (_Float16)W[(size_t)n * 512 + ks * 32 + kloc];
  }
}

// ---------------------------------------------------------------------------
// Persistent recurrence: one workgroup (32 waves) on one WGP.
// h double-buffered in LDS (2 x 64 rows x 520 halves) -> 1 barrier/step.
// xp is packed fragment-order; hseq is t-major row-major [t][b][h] (f16).
//   h(t) = tanh(xp[t] + h(t-1) @ W_hh^T)
// ---------------------------------------------------------------------------
#define HROW 520          // padded row stride in halves (conflict-free b128)
#define HBUF (64 * HROW)  // one h buffer

__global__ __launch_bounds__(1024) void rnn_recur(
    const _Float16* __restrict__ Wpk, const float* __restrict__ xp,
    _Float16* __restrict__ hseq, int T)
{
  extern __shared__ _Float16 Hs[];  // 2 * HBUF halves = 133,120 B
  const int tid    = threadIdx.x;
  const int lane   = tid & 31;
  const int wave   = tid >> 5;      // == ntile, 0..31
  const int laneHi = lane >> 4;
  const int lane15 = lane & 15;
  const int n      = wave * 16 + lane15;  // output column of this lane

  // h(-1) = 0 lives in buffer 0.
  for (int i = tid; i < HBUF; i += 1024) Hs[i] = (_Float16)0.f;
  __syncthreads();

  for (int t = 0; t < T; ++t) {
    const _Float16* rd = Hs + (t & 1) * HBUF;
    _Float16*       wr = Hs + ((t + 1) & 1) * HBUF;

    // Accumulators = packed xp slice: one 32B vector load per M-tile.
    // Issued before the fragment loads; in-order loadcnt retirement means the
    // first WMMA's fragment wait also covers them.
    v8f acc[4];
#pragma unroll
    for (int mt = 0; mt < 4; ++mt)
      acc[mt] = *(const v8f*)(xp + xp_base(t, wave, mt, laneHi, lane15));

    // Pull next step's xp slice toward the WGP: 128B per thread covers 128KB.
    if (t + 1 < T)
      __builtin_prefetch(xp + (size_t)(t + 1) * 32768 + tid * 32, 0, 1);

#pragma unroll 4
    for (int ks = 0; ks < 16; ++ks) {
      // Coalesced 32B/lane B-fragment from L2-resident packed W.
      const _Float16* bp = Wpk + (size_t)(((wave * 16 + ks) * 32) + lane) * 16;
      FragU bu;
      bu.h[0] = *(const v8h*)bp;
      bu.h[1] = *(const v8h*)(bp + 8);
      // All 4 A-fragments up front so WMMAs can overlap the LDS waits.
      v16h af[4];
#pragma unroll
      for (int mt = 0; mt < 4; ++mt)
        af[mt] = load_frag_lds(rd + (mt * 16 + lane15) * HROW + ks * 32, laneHi);
#pragma unroll
      for (int mt = 0; mt < 4; ++mt)
        acc[mt] = __builtin_amdgcn_wmma_f32_16x16x32_f16(
            false, af[mt], false, bu.v, (short)0, acc[mt], false, false);
    }

    // tanh, write h(t) to the other LDS buffer + t-major global sequence.
    _Float16* hout = hseq + (size_t)t * 64 * 512;
#pragma unroll
    for (int mt = 0; mt < 4; ++mt) {
#pragma unroll
      for (int v = 0; v < 8; ++v) {
        const int b = mt * 16 + v + 8 * laneHi;
        const _Float16 hf = (_Float16)tanhf(acc[mt][v]);
        wr[b * HROW + n] = hf;
        hout[(size_t)b * 512 + n] = hf;
      }
    }
    __syncthreads();  // h(t) complete before anyone starts step t+1
  }
}

// ---------------------------------------------------------------------------
// Final FC: out[b][c] = h2[T-1, b, :] . fc_W[c, :] + fc_b[c]   (64 x 2)
// ---------------------------------------------------------------------------
__global__ __launch_bounds__(128) void fc_kernel(
    const _Float16* __restrict__ h2, const float* __restrict__ fcW,
    const float* __restrict__ fcb, float* __restrict__ out, int T)
{
  const int tid = threadIdx.x;  // 0..127
  const int b = tid >> 1, c = tid & 1;
  const _Float16* hp = h2 + ((size_t)(T - 1) * 64 + b) * 512;
  const float* wp = fcW + c * 512;
  float s = fcb[c];
  for (int j = 0; j < 512; ++j) s += (float)hp[j] * wp[j];
  out[b * 2 + c] = s;
}

// ---------------------------------------------------------------------------
extern "C" void kernel_launch(void* const* d_in, const int* in_sizes, int n_in,
                              void* d_out, int out_size, void* d_ws, size_t ws_size,
                              hipStream_t stream) {
  const float* x    = (const float*)d_in[0];
  const float* Wih0 = (const float*)d_in[1];
  const float* Whh0 = (const float*)d_in[2];
  const float* bih0 = (const float*)d_in[3];
  const float* bhh0 = (const float*)d_in[4];
  const float* Wih1 = (const float*)d_in[5];
  const float* Whh1 = (const float*)d_in[6];
  const float* bih1 = (const float*)d_in[7];
  const float* bhh1 = (const float*)d_in[8];
  const float* fcW  = (const float*)d_in[9];
  const float* fcb  = (const float*)d_in[10];
  float* out = (float*)d_out;

  const int B = 64, T = 1024, I = 256, H = 512;
  const size_t BTH = (size_t)B * T * H;

  char* ws = (char*)d_ws;
  float*    xp  = (float*)ws;     ws += BTH * sizeof(float);      // 134 MB, packed
  _Float16* h1  = (_Float16*)ws;  ws += BTH * sizeof(_Float16);   //  67 MB, t-major
  _Float16* h2  = (_Float16*)ws;  ws += BTH * sizeof(_Float16);   //  67 MB, t-major
  _Float16* Wpk = (_Float16*)ws;                                  // 0.5 MB

  const int ldsBytes = 2 * HBUF * 2;  // 133,120 B (> 64KB default -> opt in)
  hipFuncSetAttribute((const void*)rnn_recur,
                      hipFuncAttributeMaxDynamicSharedMemorySize, ldsBytes);

  const dim3 gemmGrid(H / 64, (B * T) / 128);

  // Layer 0 (A = x, b-major source -> permuted rows)
  prepack_whh<<<64, 256, 0, stream>>>(Whh0, Wpk);
  proj_gemm<float, true><<<gemmGrid, 256, 0, stream>>>(x, Wih0, bih0, bhh0, xp,
                                                       B * T, H, I);
  rnn_recur<<<1, 1024, ldsBytes, stream>>>(Wpk, xp, h1, T);

  // Layer 1 (A = h1, already t-major)
  prepack_whh<<<64, 256, 0, stream>>>(Whh1, Wpk);
  proj_gemm<_Float16, false><<<gemmGrid, 256, 0, stream>>>(h1, Wih1, bih1, bhh1,
                                                           xp, B * T, H, H);
  rnn_recur<<<1, 1024, ldsBytes, stream>>>(Wpk, xp, h2, T);

  // Head
  fc_kernel<<<1, 128, 0, stream>>>(h2, fcW, fcb, out, T);
}
